// DeformationNetworkGcnHybridImageInput_50964081934464
// MI455X (gfx1250) — compile-verified, matrix operand order checked by reading, not run on hardware
//
#include <hip/hip_runtime.h>
#include <hip/hip_bf16.h>

typedef __attribute__((ext_vector_type(16))) _Float16 v16h;
typedef __attribute__((ext_vector_type(8)))  _Float16 v8h;
typedef __attribute__((ext_vector_type(4)))  _Float16 v4h;
typedef __attribute__((ext_vector_type(8)))  float    v8f;

#define BNUM   16
#define VNUM   2562
#define N_PTS  (BNUM * VNUM)      // 40992, multiple of 16
#define E_NUM  (16 * 7680)        // 122880 edges
#define HID    256

static __device__ __forceinline__ v16h cat8(v8h lo, v8h hi) {
    return __builtin_shufflevector(lo, hi, 0,1,2,3,4,5,6,7,8,9,10,11,12,13,14,15);
}

// ---------------------------------------------------------------------------
// WMMA GEMM:  Y = act( X @ W^T + bias )
//   X : N_PTS x K   (f16, K multiple of 32, row-major)
//   W : Nout  x K   (f16, row-major)  -> B fragment = W^T tile (K x 16)
//   Y : N_PTS x Nout (f16 or f32)
// Block = 128 threads = 4 waves. grid.y = M-tiles; grid.x covers groups of
// 8 N-tiles (each wave owns 2 N-tiles -> 2 accumulators sharing one A frag).
// The 16x32 A-tile is staged through a double-buffered LDS tile so the
// global A load for step k+32 overlaps the 2 WMMAs + B loads of step k,
// with a single barrier per K-step.
// ACT: 0=none, 1=leaky(0.01). OUTF16: 1 = write f16, 0 = write f32.
// ---------------------------------------------------------------------------
template <int ACT, int OUTF16>
__global__ void __launch_bounds__(128)
k_gemm(const _Float16* __restrict__ X, const _Float16* __restrict__ W,
       const float* __restrict__ bias, void* __restrict__ Yp,
       int K, int Nout)
{
    __shared__ __align__(16) _Float16 As[2][16 * 32];   // double-buffered A tile

    const int tid  = threadIdx.x;
    const int lane = tid & 31;
    const int wv   = tid >> 5;
    const int m0   = blockIdx.y * 16;
    const int tn0  = blockIdx.x * 8 + wv * 2;           // this wave's first N-tile
    const bool act0 = (tn0 * 16)       < Nout;
    const bool act1 = ((tn0 + 1) * 16) < Nout;
    const int n0 = act0 ? tn0 * 16       : 0;
    const int n1 = act1 ? (tn0 + 1) * 16 : 0;

    const int row = lane & 15;                 // M row (A) / N col (B) in tile
    const int klo = (lane < 16) ? 0 : 8;       // dense-WMMA f16 per-lane K map

    // staging: thread t copies halves [t*4, t*4+4) of the 16x32 tile;
    // t*4 == (t>>3)*32 + (t&7)*4 -> row r = t>>3, col c = (t&7)*4
    const _Float16* __restrict__ Xs =
        X + (size_t)(m0 + (tid >> 3)) * K + ((tid & 7) * 4);
    const int ldsoff = tid * 4;

    const _Float16* __restrict__ W0 = W + (size_t)(n0 + row) * K;
    const _Float16* __restrict__ W1 = W + (size_t)(n1 + row) * K;

    v8f acc0 = {}, acc1 = {};

    v4h st = *(const v4h*)Xs;                  // preload first A chunk
    int buf = 0;
    for (int k0 = 0; k0 < K; k0 += 32) {
        *(v4h*)(&As[buf][ldsoff]) = st;
        __syncthreads();

        // issue next A chunk early: waited on only at next iteration's store
        if (k0 + 32 < K) {
            st = *(const v4h*)(Xs + k0 + 32);
            __builtin_prefetch(W0 + k0 + 64, 0, 1);   // global_prefetch_b8
        }

        v8h alo  = *(const v8h*)(&As[buf][row * 32 + klo]);
        v8h ahi  = *(const v8h*)(&As[buf][row * 32 + klo + 16]);
        v8h b0lo = *(const v8h*)(W0 + k0 + klo);
        v8h b0hi = *(const v8h*)(W0 + k0 + klo + 16);
        v8h b1lo = *(const v8h*)(W1 + k0 + klo);
        v8h b1hi = *(const v8h*)(W1 + k0 + klo + 16);

        v16h a  = cat8(alo, ahi);
        v16h b0 = cat8(b0lo, b0hi);
        v16h b1 = cat8(b1lo, b1hi);

        acc0 = __builtin_amdgcn_wmma_f32_16x16x32_f16(
                   false, a, false, b0, (short)0, acc0, false, false);
        acc1 = __builtin_amdgcn_wmma_f32_16x16x32_f16(
                   false, a, false, b1, (short)0, acc1, false, false);
        buf ^= 1;
    }

    const int mrow = m0 + ((lane >> 4) << 3);
    if (act0) {
        const int n = n0 + row;
        const float bv = bias[n];
#pragma unroll
        for (int r = 0; r < 8; ++r) {
            float v = acc0[r] + bv;
            if (ACT == 1) v = (v > 0.f) ? v : 0.01f * v;
            const size_t o = (size_t)(mrow + r) * Nout + n;
            if (OUTF16) ((_Float16*)Yp)[o] = (_Float16)v;
            else        ((float*)Yp)[o]    = v;
        }
    }
    if (act1) {
        const int n = n1 + row;
        const float bv = bias[n];
#pragma unroll
        for (int r = 0; r < 8; ++r) {
            float v = acc1[r] + bv;
            if (ACT == 1) v = (v > 0.f) ? v : 0.01f * v;
            const size_t o = (size_t)(mrow + r) * Nout + n;
            if (OUTF16) ((_Float16*)Yp)[o] = (_Float16)v;
            else        ((float*)Yp)[o]    = v;
        }
    }
}

// ---------------------------------------------------------------------------
// Bilinear sampling of the 4 feature maps -> X0 rows (f16, stride 3872)
// cols [0,3840) = features, [3840,3843) = verts, [3843,3872) = zero pad
// One block (256 threads) per vertex.
// ---------------------------------------------------------------------------
__device__ inline void sample_map(const float* __restrict__ f, int b, int C, int R,
                                  float px, float py, int colbase,
                                  _Float16* __restrict__ xrow, int tid)
{
    const float x  = (px + 1.f) * 0.5f * (float)(R - 1);
    const float y  = (py + 1.f) * 0.5f * (float)(R - 1);
    const float x0 = floorf(x), y0 = floorf(y);

    float  w[4];
    size_t base[4];
    int k = 0;
#pragma unroll
    for (int dy = 0; dy < 2; ++dy)
#pragma unroll
        for (int dx = 0; dx < 2; ++dx) {
            const float fx = x0 + (float)dx, fy = y0 + (float)dy;
            const float wt = (1.f - fabsf(x - fx)) * (1.f - fabsf(y - fy));
            const bool valid = (fx >= 0.f) && (fx < (float)R) &&
                               (fy >= 0.f) && (fy < (float)R);
            const int xi = (int)fminf(fmaxf(fx, 0.f), (float)(R - 1));
            const int yi = (int)fminf(fmaxf(fy, 0.f), (float)(R - 1));
            w[k]    = valid ? wt : 0.f;
            base[k] = ((size_t)(b * C) * R + yi) * R + xi;
            ++k;
        }
    const size_t cs = (size_t)R * R;
    for (int c = tid; c < C; c += 256) {
        const size_t co = (size_t)c * cs;
        float v = w[0] * f[base[0] + co] + w[1] * f[base[1] + co] +
                  w[2] * f[base[2] + co] + w[3] * f[base[3] + co];
        xrow[colbase + c] = (_Float16)v;
    }
}

__global__ void __launch_bounds__(256)
k_bilinear(const float* __restrict__ f1, const float* __restrict__ f2,
           const float* __restrict__ f3, const float* __restrict__ f4,
           const float* __restrict__ av, const float* __restrict__ verts,
           _Float16* __restrict__ X0)
{
    const int row = blockIdx.x;          // 0 .. N_PTS-1
    const int b   = row / VNUM;
    const float px = av[(size_t)row * 2 + 0];
    const float py = av[(size_t)row * 2 + 1];
    _Float16* xrow = X0 + (size_t)row * 3872;

    sample_map(f1, b,  256, 56, px, py,    0, xrow, threadIdx.x);
    sample_map(f2, b,  512, 28, px, py,  256, xrow, threadIdx.x);
    sample_map(f3, b, 1024, 14, px, py,  768, xrow, threadIdx.x);
    sample_map(f4, b, 2048,  7, px, py, 1792, xrow, threadIdx.x);

    if (threadIdx.x < 32) {
        const int c = 3840 + threadIdx.x;
        const float v = (threadIdx.x < 3) ? verts[(size_t)row * 3 + threadIdx.x] : 0.f;
        xrow[c] = (_Float16)v;
    }
}

// f32 weight (Nout x Kin, row-major) -> f16 padded (Nout x Kpad)
__global__ void k_cvt_w(const float* __restrict__ w, _Float16* __restrict__ w16,
                        int kin, int kpad, int total)
{
    const int i = blockIdx.x * blockDim.x + threadIdx.x;
    if (i >= total) return;
    const int r = i / kpad, c = i % kpad;
    w16[i] = (c < kin) ? (_Float16)w[(size_t)r * kin + c] : (_Float16)0.f;
}

// hcat = [h(256) | verts(3) | 0 .. ] stride 288, f16
__global__ void k_hcat(const _Float16* __restrict__ h, const float* __restrict__ verts,
                       _Float16* __restrict__ hcat, int total)
{
    const int i = blockIdx.x * blockDim.x + threadIdx.x;
    if (i >= total) return;
    const int r = i / 288, c = i % 288;
    _Float16 v;
    if (c < 256)      v = h[(size_t)r * 256 + c];
    else if (c < 259) v = (_Float16)verts[(size_t)r * 3 + (c - 256)];
    else              v = (_Float16)0.f;
    hcat[i] = v;
}

// out[e0] += nbr[e1]; out[e1] += nbr[e0]  (per channel, float atomics)
__global__ void __launch_bounds__(256)
k_scatter(const int* __restrict__ edges, const float* __restrict__ nbr,
          float* __restrict__ out)
{
    const int e  = blockIdx.x;
    const int c  = threadIdx.x;
    const int e0 = edges[2 * e + 0];
    const int e1 = edges[2 * e + 1];
    atomicAdd(out + (size_t)e0 * HID + c, nbr[(size_t)e1 * HID + c]);
    atomicAdd(out + (size_t)e1 * HID + c, nbr[(size_t)e0 * HID + c]);
}

__global__ void k_relu_cvt(const float* __restrict__ src, _Float16* __restrict__ dst, int n)
{
    const int i = blockIdx.x * blockDim.x + threadIdx.x;
    if (i < n) { const float v = src[i]; dst[i] = (_Float16)(v > 0.f ? v : 0.f); }
}

// final layer: (N x 32) f16 @ (3 x 32)^T f32 + bias -> (N x 3) f32
__global__ void k_final(const _Float16* __restrict__ h, const float* __restrict__ w,
                        const float* __restrict__ bias, float* __restrict__ out)
{
    const int i = blockIdx.x * blockDim.x + threadIdx.x;
    if (i >= N_PTS) return;
    const _Float16* hr = h + (size_t)i * 32;
#pragma unroll
    for (int c = 0; c < 3; ++c) {
        float acc = bias[c];
#pragma unroll
        for (int k = 0; k < 32; ++k) acc += (float)hr[k] * w[c * 32 + k];
        out[(size_t)i * 3 + c] = acc;
    }
}

// ---------------------------------------------------------------------------
extern "C" void kernel_launch(void* const* d_in, const int* in_sizes, int n_in,
                              void* d_out, int out_size, void* d_ws, size_t ws_size,
                              hipStream_t stream)
{
    (void)in_sizes; (void)n_in; (void)out_size; (void)ws_size;

    const float* feat1 = (const float*)d_in[0];
    const float* feat2 = (const float*)d_in[1];
    const float* feat3 = (const float*)d_in[2];
    const float* feat4 = (const float*)d_in[3];
    const float* av    = (const float*)d_in[4];
    const float* verts = (const float*)d_in[5];
    const int*   edges = (const int*)d_in[6];

    const float *w1[4], *b1[4];
    for (int i = 0; i < 4; ++i) { w1[i] = (const float*)d_in[7 + 2*i]; b1[i] = (const float*)d_in[8 + 2*i]; }
    const float *gw0[10], *gb0[10], *gw1[10], *gb1[10];
    for (int i = 0; i < 10; ++i) {
        gw0[i] = (const float*)d_in[15 + 4*i + 0];
        gb0[i] = (const float*)d_in[15 + 4*i + 1];
        gw1[i] = (const float*)d_in[15 + 4*i + 2];
        gb1[i] = (const float*)d_in[15 + 4*i + 3];
    }
    const float *w3[4], *b3[4];
    for (int i = 0; i < 4; ++i) { w3[i] = (const float*)d_in[55 + 2*i]; b3[i] = (const float*)d_in[56 + 2*i]; }

    // ---- workspace carve ----
    char* ws = (char*)d_ws;
    size_t off = 0;
    auto alloc = [&](size_t bytes) -> char* {
        char* p = ws + off;
        off += (bytes + 255) & ~(size_t)255;
        return p;
    };
    const size_t N = N_PTS;
    _Float16* X0   = (_Float16*)alloc(N * 3872 * 2);
    _Float16* Y1   = (_Float16*)alloc(N * 512 * 2);
    _Float16* Y2   = (_Float16*)alloc(N * 512 * 2);
    _Float16* Y3   = (_Float16*)alloc(N * 256 * 2);
    _Float16* H16  = (_Float16*)alloc(N * 256 * 2);
    _Float16* Hcat = (_Float16*)alloc(N * 288 * 2);
    float*    OUT32 = (float*)alloc(N * 256 * 4);
    float*    NBR32 = (float*)alloc(N * 256 * 4);
    _Float16* M1   = (_Float16*)alloc(N * 128 * 2);
    _Float16* M2   = (_Float16*)alloc(N * 64 * 2);
    _Float16* M3   = (_Float16*)alloc(N * 32 * 2);

    _Float16* mw16[4];
    const int m1_nout[4] = {512, 512, 256, 256};
    const int m1_kin [4] = {3843, 512, 512, 256};
    const int m1_kpad[4] = {3872, 512, 512, 256};
    for (int i = 0; i < 4; ++i) mw16[i] = (_Float16*)alloc((size_t)m1_nout[i] * m1_kpad[i] * 2);

    _Float16 *g16a[10], *g16b[10];
    for (int i = 0; i < 10; ++i) {
        const int kp = (i == 0) ? 288 : 256;
        g16a[i] = (_Float16*)alloc((size_t)256 * kp * 2);
        g16b[i] = (_Float16*)alloc((size_t)256 * kp * 2);
    }
    _Float16* m316[3];
    const int m3_nout[3] = {128, 64, 32};
    const int m3_k   [3] = {256, 128, 64};
    for (int i = 0; i < 3; ++i) m316[i] = (_Float16*)alloc((size_t)m3_nout[i] * m3_k[i] * 2);

    // ---- helpers ----
    auto cvt = [&](const float* w, _Float16* w16, int nout, int kin, int kpad) {
        const int total = nout * kpad;
        k_cvt_w<<<(total + 255) / 256, 256, 0, stream>>>(w, w16, kin, kpad, total);
    };
    auto gemm16 = [&](const _Float16* X, const _Float16* W, const float* b,
                      _Float16* Y, int K, int Nout) {
        dim3 g((unsigned)((Nout / 16 + 7) / 8), (unsigned)(N_PTS / 16));
        k_gemm<1, 1><<<g, 128, 0, stream>>>(X, W, b, (void*)Y, K, Nout);
    };
    auto gemm32 = [&](const _Float16* X, const _Float16* W, const float* b,
                      float* Y, int K, int Nout) {
        dim3 g((unsigned)((Nout / 16 + 7) / 8), (unsigned)(N_PTS / 16));
        k_gemm<0, 0><<<g, 128, 0, stream>>>(X, W, b, (void*)Y, K, Nout);
    };

    // ---- weight conversion (f32 -> padded f16) ----
    for (int i = 0; i < 4; ++i) cvt(w1[i], mw16[i], m1_nout[i], m1_kin[i], m1_kpad[i]);
    for (int i = 0; i < 10; ++i) {
        const int kin = (i == 0) ? 259 : 256, kp = (i == 0) ? 288 : 256;
        cvt(gw0[i], g16a[i], 256, kin, kp);
        cvt(gw1[i], g16b[i], 256, kin, kp);
    }
    for (int i = 0; i < 3; ++i) cvt(w3[i], m316[i], m3_nout[i], m3_k[i], m3_k[i]);

    // ---- stage 1: bilinear sampling -> X0 ----
    k_bilinear<<<N_PTS, 256, 0, stream>>>(feat1, feat2, feat3, feat4, av, verts, X0);

    // ---- stage 2: MLP1 (leaky relu, f16 activations) ----
    gemm16(X0, mw16[0], b1[0], Y1, 3872, 512);
    gemm16(Y1, mw16[1], b1[1], Y2,  512, 512);
    gemm16(Y2, mw16[2], b1[2], Y3,  512, 256);
    gemm16(Y3, mw16[3], b1[3], H16, 256, 256);

    // ---- stage 3: GCN layers ----
    {
        const int tot = N_PTS * 288;
        k_hcat<<<(tot + 255) / 256, 256, 0, stream>>>(H16, verts, Hcat, tot);
    }
    for (int i = 0; i < 10; ++i) {
        const _Float16* xin = (i == 0) ? Hcat : H16;
        const int K = (i == 0) ? 288 : 256;
        gemm32(xin, g16a[i], gb0[i], OUT32, K, 256);
        gemm32(xin, g16b[i], gb1[i], NBR32, K, 256);
        k_scatter<<<E_NUM, 256, 0, stream>>>(edges, NBR32, OUT32);
        const int n = N_PTS * 256;
        k_relu_cvt<<<(n + 255) / 256, 256, 0, stream>>>(OUT32, H16, n);
    }

    // ---- stage 4: MLP3 ----
    gemm16(H16, m316[0], b3[0], M1, 256, 128);
    gemm16(M1,  m316[1], b3[1], M2, 128, 64);
    gemm16(M2,  m316[2], b3[2], M3,  64, 32);
    k_final<<<(N_PTS + 255) / 256, 256, 0, stream>>>(M3, w3[3], b3[3], (float*)d_out);
}
